// CoarseMatching_56281251447263
// MI455X (gfx1250) — matrix operand order repaired and checked by reading, not compiled, not run on hardware
//
#include <hip/hip_runtime.h>
#include <stdint.h>

// ---------------- problem constants (from reference setup_inputs) -----------
#define NB 4
#define LL 4800
#define SS 4800
#define CC 256
#define CH 16                     // L-chunks for column reductions
#define ROWS_PER_CHUNK (LL / CH)  // 300
#define THRV 0.2f
#define BORD 2
#define INV_TEMP 10.0f            // 1/0.1

typedef __attribute__((ext_vector_type(16))) __bf16 v16bf;
typedef __attribute__((ext_vector_type(8)))  __bf16 v8bf;
typedef __attribute__((ext_vector_type(8)))  float  v8f;

// ---------------- 1) fp32 -> bf16 (RNE) with 1/16 scale folded in ----------
__global__ __launch_bounds__(256)
void cvt_scale_bf16(const float* __restrict__ in, uint16_t* __restrict__ out, int n4) {
  int i = blockIdx.x * 256 + threadIdx.x;
  if (i >= n4) return;
  const float4 f = ((const float4*)in)[i];
  float v[4] = {f.x, f.y, f.z, f.w};
  uint16_t r[4];
#pragma unroll
  for (int k = 0; k < 4; ++k) {
    uint32_t x = __float_as_uint(v[k] * 0.0625f);   // * 1/sqrt(256) per operand
    uint32_t rr = x + 0x7fffu + ((x >> 16) & 1u);   // round-to-nearest-even
    r[k] = (uint16_t)(rr >> 16);
  }
  ushort4 o; o.x = r[0]; o.y = r[1]; o.z = r[2]; o.w = r[3];
  ((ushort4*)out)[i] = o;
}

// ---------------- 2) sim GEMM via v_wmma_f32_16x16x32_bf16 ------------------
// Block = 256 threads = 8 waves. Block tile 64(L) x 64(S).
// Wave w: rows [16*(w>>1), +16), cols [32*(w&1), +32) => two 16x16 accumulators.
__global__ __launch_bounds__(256)
void sim_gemm(const __bf16* __restrict__ A, const __bf16* __restrict__ B,
              float* __restrict__ sim) {
  const int batch = blockIdx.z;
  const int l0 = blockIdx.y * 64;
  const int s0 = blockIdx.x * 64;
  const int tid = threadIdx.x;
  const int wave = tid >> 5;
  const int lane = tid & 31;
  const int hs = lane >> 4;      // 0 | 1 (half-wave select)
  const int ln = lane & 15;

  const int rowBase = l0 + 16 * (wave >> 1);
  const int colBase = s0 + 32 * (wave & 1);

  // A fragment addressing in contiguous K order (k = 16h + s, same as B):
  // lane<16 holds row m=ln, K {k0..k0+15}; lane>=16 offsets by 16.
  const __bf16* aPtr = A + ((size_t)batch * LL + rowBase + ln) * CC + hs * 16;
  // B fragment: lane holds column n=ln, 16 contiguous K halves, half-wave K offset 16.
  const __bf16* bPtr0 = B + ((size_t)batch * SS + colBase + ln) * CC + hs * 16;
  const __bf16* bPtr1 = bPtr0 + (size_t)16 * CC;

  v8f acc0 = {};
  v8f acc1 = {};
#pragma unroll
  for (int k0 = 0; k0 < CC; k0 += 32) {
    v8bf alo = *(const v8bf*)(aPtr + k0);
    v8bf ahi = *(const v8bf*)(aPtr + k0 + 8);
    v16bf a = __builtin_shufflevector(alo, ahi, 0,1,2,3,4,5,6,7,8,9,10,11,12,13,14,15);
    v16bf b0 = *(const v16bf*)(bPtr0 + k0);
    v16bf b1 = *(const v16bf*)(bPtr1 + k0);
    acc0 = __builtin_amdgcn_wmma_f32_16x16x32_bf16(false, a, false, b0, (short)0, acc0, false, false);
    acc1 = __builtin_amdgcn_wmma_f32_16x16x32_bf16(false, a, false, b1, (short)0, acc1, false, false);
  }

  float* out = sim + (size_t)batch * LL * SS;
#pragma unroll
  for (int r = 0; r < 8; ++r) {
    const int row = rowBase + r + hs * 8;            // C/D layout: VGPR r -> M=r / r+8
    __builtin_nontemporal_store(acc0[r], &out[(size_t)row * SS + (colBase + ln)]);
    __builtin_nontemporal_store(acc1[r], &out[(size_t)row * SS + (colBase + 16 + ln)]);
  }
}

// ---------------- helpers: online softmax pair combine ----------------------
__device__ __forceinline__ void online_combine(float& m, float& s, float m2, float s2) {
  float mm = fmaxf(m, m2);
  s = s * __expf(m - mm) + s2 * __expf(m2 - mm);
  m = mm;
}

// ---------------- 3a) per-row softmax stats (max, 1/sumexp) -----------------
__global__ __launch_bounds__(256)
void row_stats(const float* __restrict__ sim, float* __restrict__ rowmax,
               float* __restrict__ rowsuminv) {
  const int row = blockIdx.x;            // n*L + l
  const float* p = sim + (size_t)row * SS;
  float m = -INFINITY, s = 0.f;
  for (int i = threadIdx.x; i < SS; i += 256) {
    float x = __builtin_nontemporal_load(&p[i]) * INV_TEMP;
    if (x > m) { s *= __expf(m - x); m = x; }
    s += __expf(x - m);
  }
  __shared__ float shm[256], shs[256];
  shm[threadIdx.x] = m; shs[threadIdx.x] = s;
  __syncthreads();
  for (int off = 128; off > 0; off >>= 1) {
    if (threadIdx.x < off) {
      float m1 = shm[threadIdx.x], s1 = shs[threadIdx.x];
      online_combine(m1, s1, shm[threadIdx.x + off], shs[threadIdx.x + off]);
      shm[threadIdx.x] = m1; shs[threadIdx.x] = s1;
    }
    __syncthreads();
  }
  if (threadIdx.x == 0) { rowmax[row] = shm[0]; rowsuminv[row] = 1.0f / shs[0]; }
}

// ---------------- 3b) per-column stats: chunked online pass -----------------
__global__ __launch_bounds__(256)
void col_stats_part(const float* __restrict__ sim, float* __restrict__ pm,
                    float* __restrict__ ps) {
  const int col = blockIdx.x * 256 + threadIdx.x;
  const int chunk = blockIdx.y, batch = blockIdx.z;
  if (col >= SS) return;
  const float* p = sim + ((size_t)batch * LL + (size_t)chunk * ROWS_PER_CHUNK) * SS + col;
  float m = -INFINITY, s = 0.f;
  for (int i = 0; i < ROWS_PER_CHUNK; ++i) {
    float x = __builtin_nontemporal_load(&p[(size_t)i * SS]) * INV_TEMP;
    if (x > m) { s *= __expf(m - x); m = x; }
    s += __expf(x - m);
  }
  const size_t o = ((size_t)batch * CH + chunk) * SS + col;
  pm[o] = m; ps[o] = s;
}

__global__ __launch_bounds__(256)
void col_stats_comb(const float* __restrict__ pm, const float* __restrict__ ps,
                    float* __restrict__ colmax, float* __restrict__ colsuminv) {
  const int col = blockIdx.x * 256 + threadIdx.x;
  const int batch = blockIdx.y;
  if (col >= SS) return;
  float m = -INFINITY, s = 0.f;
  for (int c = 0; c < CH; ++c) {
    const size_t o = ((size_t)batch * CH + c) * SS + col;
    online_combine(m, s, pm[o], ps[o]);
  }
  colmax[(size_t)batch * SS + col] = m;
  colsuminv[(size_t)batch * SS + col] = 1.0f / s;
}

// ---------------- 4) conf = softmax_L * softmax_S; track per-row conf max ---
__global__ __launch_bounds__(256)
void conf_rows(const float* __restrict__ sim, const float* __restrict__ rowmax,
               const float* __restrict__ rowsuminv, const float* __restrict__ colmax,
               const float* __restrict__ colsuminv, float* __restrict__ conf,
               float* __restrict__ rowcmax) {
  const int row = blockIdx.x;            // n*L + l
  const int batch = row / LL;
  const float* ps_ = sim + (size_t)row * SS;
  float* pc = conf + (size_t)row * SS;
  const float rm = rowmax[row], ri = rowsuminv[row];
  const float* cm = colmax + (size_t)batch * SS;
  const float* ci = colsuminv + (size_t)batch * SS;
  float cmax = 0.f;
  for (int s = threadIdx.x; s < SS; s += 256) {
    float st = __builtin_nontemporal_load(&ps_[s]) * INV_TEMP;
    float c = __expf(2.0f * st - rm - cm[s]) * ri * ci[s];  // exp(a)*exp(b)=exp(a+b)
    __builtin_nontemporal_store(c, &pc[s]);
    cmax = fmaxf(cmax, c);
  }
  __shared__ float sh[256];
  sh[threadIdx.x] = cmax;
  __syncthreads();
  for (int off = 128; off > 0; off >>= 1) {
    if (threadIdx.x < off) sh[threadIdx.x] = fmaxf(sh[threadIdx.x], sh[threadIdx.x + off]);
    __syncthreads();
  }
  if (threadIdx.x == 0) rowcmax[row] = sh[0];
}

// ---------------- 5) per-column conf max (chunk + combine) ------------------
__global__ __launch_bounds__(256)
void col_cmax_part(const float* __restrict__ conf, float* __restrict__ pm) {
  const int col = blockIdx.x * 256 + threadIdx.x;
  const int chunk = blockIdx.y, batch = blockIdx.z;
  if (col >= SS) return;
  const float* p = conf + ((size_t)batch * LL + (size_t)chunk * ROWS_PER_CHUNK) * SS + col;
  float m = 0.f;
  for (int i = 0; i < ROWS_PER_CHUNK; ++i)
    m = fmaxf(m, __builtin_nontemporal_load(&p[(size_t)i * SS]));
  pm[((size_t)batch * CH + chunk) * SS + col] = m;
}

__global__ __launch_bounds__(256)
void col_cmax_comb(const float* __restrict__ pm, float* __restrict__ colcmax) {
  const int col = blockIdx.x * 256 + threadIdx.x;
  const int batch = blockIdx.y;
  if (col >= SS) return;
  float m = 0.f;
  for (int c = 0; c < CH; ++c) m = fmaxf(m, pm[((size_t)batch * CH + c) * SS + col]);
  colcmax[(size_t)batch * SS + col] = m;
}

// ---------------- 6) threshold + border + mutual-NN + argmax ----------------
__global__ __launch_bounds__(256)
void finalize(const float* __restrict__ conf, const float* __restrict__ rowcmax,
              const float* __restrict__ colcmax,
              const int* __restrict__ h0c_, const int* __restrict__ w0c_,
              const int* __restrict__ h1c_, const int* __restrict__ w1c_,
              const int* __restrict__ h0i_,
              float* __restrict__ mask_v, float* __restrict__ all_j,
              float* __restrict__ mconf, float* __restrict__ mk0,
              float* __restrict__ mk1) {
  const int row = blockIdx.x;            // n*L + l
  const int n = row / LL, l = row % LL;
  const int h0c = *h0c_, w0c = *w0c_, h1c = *h1c_, w1c = *w1c_, h0i = *h0i_;
  const float scale = (float)h0i / (float)h0c;
  const int iy = l / w0c, ix = l % w0c;
  const bool vrow = (iy >= BORD) && (iy < h0c - BORD) && (ix >= BORD) && (ix < w0c - BORD);
  const float rcm = rowcmax[row];
  const float* pc = conf + (size_t)row * SS;
  const float* ccm = colcmax + (size_t)n * SS;

  int best = 0x7fffffff;
  if (vrow) {
    for (int s = threadIdx.x; s < SS; s += 256) {
      float c = pc[s];
      if (c > THRV && c == rcm && c == ccm[s]) {
        int sy = s / w1c, sx = s % w1c;
        if (sy >= BORD && sy < h1c - BORD && sx >= BORD && sx < w1c - BORD)
          best = min(best, s);    // jnp.argmax(maskf) = first true index
      }
    }
  }
  __shared__ int sh[256];
  sh[threadIdx.x] = best;
  __syncthreads();
  for (int off = 128; off > 0; off >>= 1) {
    if (threadIdx.x < off) sh[threadIdx.x] = min(sh[threadIdx.x], sh[threadIdx.x + off]);
    __syncthreads();
  }
  if (threadIdx.x == 0) {
    int j = sh[0];
    bool found = (j != 0x7fffffff);
    if (!found) j = 0;                    // argmax of all-zero maskf == 0
    mask_v[row] = found ? 1.f : 0.f;
    all_j[row]  = (float)j;
    mconf[row]  = found ? pc[j] : 0.f;
    mk1[(size_t)row * 2 + 0] = (float)(j % w1c) * scale;
    mk1[(size_t)row * 2 + 1] = (float)(j / w1c) * scale;
    if (n == 0) {
      mk0[(size_t)l * 2 + 0] = (float)ix * scale;
      mk0[(size_t)l * 2 + 1] = (float)iy * scale;
    }
  }
}

// ---------------- launcher --------------------------------------------------
extern "C" void kernel_launch(void* const* d_in, const int* in_sizes, int n_in,
                              void* d_out, int out_size, void* d_ws, size_t ws_size,
                              hipStream_t stream) {
  (void)in_sizes; (void)n_in; (void)out_size; (void)ws_size;
  const float* feat0 = (const float*)d_in[0];
  const float* feat1 = (const float*)d_in[1];
  const int* h0c = (const int*)d_in[2];
  const int* w0c = (const int*)d_in[3];
  const int* h1c = (const int*)d_in[4];
  const int* w1c = (const int*)d_in[5];
  const int* h0i = (const int*)d_in[6];

  const size_t NLC = (size_t)NB * LL * CC;       // 4,915,200
  const size_t NLS = (size_t)NB * LL * SS;       // 92,160,000
  const size_t NL  = (size_t)NB * LL;            // 19,200

  // ---- workspace carve-up (all regions fully written before read) ----
  char* ws = (char*)d_ws;
  uint16_t* abf       = (uint16_t*)ws;                 ws += NLC * 2;
  uint16_t* bbf       = (uint16_t*)ws;                 ws += NLC * 2;
  float* rowmax       = (float*)ws;                    ws += NL * 4;
  float* rowsuminv    = (float*)ws;                    ws += NL * 4;
  float* colmax       = (float*)ws;                    ws += (size_t)NB * SS * 4;
  float* colsuminv    = (float*)ws;                    ws += (size_t)NB * SS * 4;
  float* rowcmax      = (float*)ws;                    ws += NL * 4;
  float* colcmax      = (float*)ws;                    ws += (size_t)NB * SS * 4;
  float* pm           = (float*)ws;                    ws += (size_t)NB * CH * SS * 4;
  float* ps           = (float*)ws;                    /* end */

  // ---- output carve-up (return order) ----
  float* conf   = (float*)d_out;
  float* sim    = conf + NLS;
  float* mask_v = sim + NLS;
  float* all_j  = mask_v + NL;
  float* mconf  = all_j + NL;
  float* mk0    = mconf + NL;
  float* mk1    = mk0 + (size_t)LL * 2;

  // 1) bf16 conversion (vectorized x4)
  {
    int n4 = (int)(NLC / 4);
    dim3 g((n4 + 255) / 256);
    cvt_scale_bf16<<<g, 256, 0, stream>>>(feat0, abf, n4);
    cvt_scale_bf16<<<g, 256, 0, stream>>>(feat1, bbf, n4);
  }
  // 2) WMMA GEMM: 75x75 tiles of 64x64 per batch
  sim_gemm<<<dim3(SS / 64, LL / 64, NB), 256, 0, stream>>>(
      (const __bf16*)abf, (const __bf16*)bbf, sim);
  // 3) softmax statistics
  row_stats<<<dim3((unsigned)NL), 256, 0, stream>>>(sim, rowmax, rowsuminv);
  col_stats_part<<<dim3((SS + 255) / 256, CH, NB), 256, 0, stream>>>(sim, pm, ps);
  col_stats_comb<<<dim3((SS + 255) / 256, NB), 256, 0, stream>>>(pm, ps, colmax, colsuminv);
  // 4) conf + per-row conf max
  conf_rows<<<dim3((unsigned)NL), 256, 0, stream>>>(sim, rowmax, rowsuminv, colmax,
                                                    colsuminv, conf, rowcmax);
  // 5) per-column conf max
  col_cmax_part<<<dim3((SS + 255) / 256, CH, NB), 256, 0, stream>>>(conf, pm);
  col_cmax_comb<<<dim3((SS + 255) / 256, NB), 256, 0, stream>>>(pm, colcmax);
  // 6) matching + keypoints
  finalize<<<dim3((unsigned)NL), 256, 0, stream>>>(conf, rowcmax, colcmax, h0c, w0c, h1c,
                                                   w1c, h0i, mask_v, all_j, mconf, mk0, mk1);
}
